// AdaptiveStateSelector_57878979281522
// MI455X (gfx1250) — compile-verified, matrix operand order checked by reading, not run on hardware
//
#include <hip/hip_runtime.h>
#include <hip/hip_bf16.h>

typedef __attribute__((ext_vector_type(16))) _Float16 v16h;
typedef __attribute__((ext_vector_type(8)))  float    v8f;

// ---------------- problem constants ----------------
static constexpr int BB   = 2;      // batch
static constexpr int HH   = 48;
static constexpr int WW   = 48;
static constexpr int LL   = HH * WW;        // 2304
static constexpr int CC   = 96;     // model dim
static constexpr int DD   = 192;    // inner dim (expand 2)
static constexpr int RR   = 6;      // dt_rank = ceil(96/16)
static constexpr int KK   = 4;      // scan directions
static constexpr int NCHUNK = 48;   // chunks for parallel scan
static constexpr int CLEN   = 48;   // L / NCHUNK

// =====================================================================
// Gate: pooled mean -> MLP (96->24->3) -> softmax
// =====================================================================
__global__ void gate_kernel(const float* __restrict__ x,
                            const float* __restrict__ w1, const float* __restrict__ b1,
                            const float* __restrict__ w2, const float* __restrict__ b2,
                            float* __restrict__ gates) {
    const int MID = 24;
    __shared__ float pooled[BB * CC];
    __shared__ float hmid[BB * MID];
    __shared__ float logits[BB * 3];
    int t = threadIdx.x;
    if (t < BB * CC) {
        int b = t / CC, cc = t % CC;
        const float* xp = x + (size_t)b * LL * CC + cc;
        float s = 0.f;
        for (int l = 0; l < LL; ++l) s += xp[(size_t)l * CC];
        pooled[t] = s / (float)LL;
    }
    __syncthreads();
    if (t < BB * MID) {
        int b = t / MID, j = t % MID;
        float s = b1[j];
        for (int cc = 0; cc < CC; ++cc) s += pooled[b * CC + cc] * w1[j * CC + cc];
        hmid[t] = s > 0.f ? s : 0.f;
    }
    __syncthreads();
    if (t < BB * 3) {
        int b = t / 3, o = t % 3;
        float s = b2[o];
        for (int j = 0; j < MID; ++j) s += hmid[b * MID + j] * w2[o * MID + j];
        logits[t] = s;
    }
    __syncthreads();
    if (t < BB) {
        float m = fmaxf(logits[t * 3], fmaxf(logits[t * 3 + 1], logits[t * 3 + 2]));
        float e0 = __expf(logits[t * 3 + 0] - m);
        float e1 = __expf(logits[t * 3 + 1] - m);
        float e2 = __expf(logits[t * 3 + 2] - m);
        float inv = 1.f / (e0 + e1 + e2);
        gates[t * 3 + 0] = e0 * inv;
        gates[t * 3 + 1] = e1 * inv;
        gates[t * 3 + 2] = e2 * inv;
    }
}

__global__ void zero_kernel(float* __restrict__ p, int nelem) {
    int t = blockIdx.x * blockDim.x + threadIdx.x;
    if (t < nelem) p[t] = 0.f;
}

// =====================================================================
// Generic WMMA GEMM: O(M x N) = A(M x K) * W(N x K)^T   (f16 WMMA, f32 acc)
//   - one wave32 per 16x16 output tile; K (template) multiple of 32,
//     fully unrolled -> K/32 static v_wmma per instantiation
//   - RAGGED: N not multiple of 16 -> clamped unconditional loads + cndmask
//     (no EXEC-divergent loads), guarded stores
//   - ACCUM: O += gate[b][branch] * A*W^T (out-proj fusion)
//   - batched: A/O stride per blockIdx.z, W stride per (z % wMod)
// Fragment layouts per CDNA5 ISA 7.12.2 (16-bit A 16x32; 32-bit C/D 16x16).
// =====================================================================
template <int K, bool RAGGED, bool ACCUM>
__global__ void wmma_gemm_xwT(const float* __restrict__ Abase,
                              const float* __restrict__ Wbase,
                              float* __restrict__ Obase,
                              int N,
                              long sA, long sW, long sO, int wMod,
                              const float* __restrict__ gates, int branch) {
    int bz = blockIdx.z;
    const float* Ap = Abase + (size_t)bz * sA;
    const float* Wp = Wbase + (size_t)(bz % wMod) * sW;
    float*       Op = Obase + (size_t)bz * sO;

    int lane = threadIdx.x & 31;
    int half = lane >> 4;                      // K-half selector for A/B frags
    int mrow = blockIdx.y * 16 + (lane & 15);  // A row (same M for both halves)
    int nrow = blockIdx.x * 16 + (lane & 15);  // W row (output channel)
    bool nok = true;
    int  nr  = nrow;
    if (RAGGED) {                              // clamp row, zero later via select
        nok = (nrow < N);
        nr  = nok ? nrow : (N - 1);
    }
    const float* arow = Ap + (size_t)mrow * K;
    const float* wrow = Wp + (size_t)nr * K;

    v8f acc = {};
#pragma unroll
    for (int k0 = 0; k0 < K; k0 += 32) {
        v16h af, bf;
#pragma unroll
        for (int v = 0; v < 8; ++v) {
            // VGPR v holds K = k0 + 16*(v>>2) + 8*half + 2*(v&3) .. +1
            int kk = k0 + ((v >> 2) << 4) + (half << 3) + ((v & 3) << 1);
            float a0 = arow[kk];
            float a1 = arow[kk + 1];
            af[2 * v]     = (_Float16)a0;
            af[2 * v + 1] = (_Float16)a1;
            float b0 = wrow[kk];
            float b1 = wrow[kk + 1];
            if (RAGGED) {                      // v_cndmask, no divergent loads
                b0 = nok ? b0 : 0.f;
                b1 = nok ? b1 : 0.f;
            }
            bf[2 * v]     = (_Float16)b0;
            bf[2 * v + 1] = (_Float16)b1;
        }
        // (neg_a, A, neg_b, B, c_mod, C, reuse_a, reuse_b)
        acc = __builtin_amdgcn_wmma_f32_16x16x32_f16(false, af, false, bf,
                                                     (short)0, acc, false, false);
    }

    int col = blockIdx.x * 16 + (lane & 15);
    if (!RAGGED || col < N) {
        float g = 1.f;
        if (ACCUM) {
            int b = (blockIdx.y * 16) / LL;    // 16-row tile never straddles batches
            g = gates[b * 3 + branch];
        }
#pragma unroll
        for (int v = 0; v < 8; ++v) {
            int row = blockIdx.y * 16 + v + (half << 3);  // C/D layout: M = v + 8*half
            size_t oi = (size_t)row * N + col;
            Op[oi] = ACCUM ? (Op[oi] + g * acc[v]) : acc[v];
        }
    }
}

// =====================================================================
// Depthwise 3x3 conv (SAME) + bias + SiLU + 4-direction cross-scan scatter
//   xz: (B, L, 384) rows; xi = channels [0,192)
//   xsT: (B, 4, L, 192)
// =====================================================================
__global__ void conv_silu_scatter(const float* __restrict__ xz,
                                  const float* __restrict__ cw,
                                  const float* __restrict__ cb,
                                  float* __restrict__ xsT) {
    int t = blockIdx.x * blockDim.x + threadIdx.x;
    if (t >= BB * DD * LL) return;
    int w = t % WW;
    int rest = t / WW;
    int h = rest % HH; rest /= HH;
    int dd = rest % DD;
    int b  = rest / DD;

    float s = cb[dd];
#pragma unroll
    for (int dh = -1; dh <= 1; ++dh) {
#pragma unroll
        for (int dw = -1; dw <= 1; ++dw) {
            int hh = h + dh, ww = w + dw;
            if (hh >= 0 && hh < HH && ww >= 0 && ww < WW) {
                float xi = xz[((size_t)b * LL + (size_t)(hh * WW + ww)) * (2 * DD) + dd];
                s += xi * cw[dd * 9 + (dh + 1) * 3 + (dw + 1)];
            }
        }
    }
    float v = s / (1.f + __expf(-s));          // SiLU

    int l  = h * WW + w;
    int lt = w * HH + h;                       // spatial transpose index
    size_t base = (size_t)b * KK * LL * DD;
    xsT[base + (size_t)l * DD + dd]                          = v;  // k=0
    xsT[base + (size_t)(LL + lt) * DD + dd]                  = v;  // k=1 (transposed)
    xsT[base + (size_t)(2 * LL + (LL - 1 - l)) * DD + dd]    = v;  // k=2 (flip)
    xsT[base + (size_t)(3 * LL + (LL - 1 - lt)) * DD + dd]   = v;  // k=3 (flip+transp)
}

// =====================================================================
// delta = softplus(dts @ dt_w^T + dt_b), K = r = 6 (scalar, tiny K)
//   xdbl: (B*4, L, c) rows; deltaT: (B*4, L, 192)
// =====================================================================
__global__ void dt_kernel(const float* __restrict__ xdbl,
                          const float* __restrict__ dt_w,
                          const float* __restrict__ dt_b,
                          float* __restrict__ deltaT, int c) {
    int t = blockIdx.x * blockDim.x + threadIdx.x;
    if (t >= BB * KK * LL * DD) return;
    int dd = t % DD;
    int rest = t / DD;
    int l  = rest % LL;
    int bk = rest / LL;
    int k  = bk & 3;
    const float* row = xdbl + ((size_t)bk * LL + l) * c;
    const float* wv  = dt_w + ((size_t)k * DD + dd) * RR;
    float s = dt_b[k * DD + dd];
#pragma unroll
    for (int j = 0; j < RR; ++j) s += row[j] * wv[j];
    deltaT[t] = (s > 20.f) ? s : log1pf(__expf(s));
}

__global__ void abar_kernel(const float* __restrict__ A_logs,
                            float* __restrict__ Abar, int nelem) {
    int t = blockIdx.x * blockDim.x + threadIdx.x;
    if (t < nelem) Abar[t] = -__expf(A_logs[t]);
}

// =====================================================================
// Chunked (associative) selective scan, 3 phases.
//   h_l = exp(dt*A)*h_{l-1} + dt*u*B_l ;  y_l = <h_l, C_l> + u*D
// Phase A: per-chunk local scan -> (prod dA, partial h) per (chunk,bk,dd,i)
// Phase B: 48-step scan over chunk summaries -> incoming state per chunk
// Phase C: replay each chunk with incoming state, emit y.
// chunk buffers layout: [chunk][bk][dd][i] so Phase B threads are coalesced.
// =====================================================================
template <int N>
__global__ void scan_phaseA(const float* __restrict__ xsT,
                            const float* __restrict__ deltaT,
                            const float* __restrict__ xdbl,
                            const float* __restrict__ Abar,
                            float* __restrict__ chunkA,
                            float* __restrict__ chunkH) {
    const int c = RR + 2 * N;
    int chunk = blockIdx.x;
    int bk    = blockIdx.y;
    int dd    = threadIdx.x;
    int k     = bk & 3;

    float Areg[N], ap[N], h[N];
#pragma unroll
    for (int i = 0; i < N; ++i) {
        Areg[i] = Abar[((size_t)k * DD + dd) * N + i];
        ap[i] = 1.f;
        h[i] = 0.f;
    }
    const float* dtp = deltaT + (size_t)bk * LL * DD;
    const float* up  = xsT    + (size_t)bk * LL * DD;
    const float* xr  = xdbl   + (size_t)bk * LL * c;

    int l0 = chunk * CLEN;
    for (int l = l0; l < l0 + CLEN; ++l) {
        float dt = dtp[(size_t)l * DD + dd];
        float u  = up[(size_t)l * DD + dd];
        float du = dt * u;
        const float* row = xr + (size_t)l * c;
#pragma unroll
        for (int i = 0; i < N; ++i) {
            float dA = __expf(dt * Areg[i]);
            h[i]  = dA * h[i] + du * row[RR + i];
            ap[i] *= dA;
        }
    }
    size_t base = ((size_t)(chunk * (BB * KK) + bk) * DD + dd) * N;
#pragma unroll
    for (int i = 0; i < N; ++i) {
        chunkA[base + i] = ap[i];
        chunkH[base + i] = h[i];
    }
}

__global__ void scan_phaseB(const float* __restrict__ chunkA,
                            float* __restrict__ chunkH, int total) {
    int t = blockIdx.x * blockDim.x + threadIdx.x;
    if (t >= total) return;
    float h = 0.f;
    for (int ch = 0; ch < NCHUNK; ++ch) {
        size_t idx = (size_t)ch * total + t;
        float a = chunkA[idx];
        float v = chunkH[idx];
        chunkH[idx] = h;          // in-place: becomes incoming state for chunk
        h = a * h + v;
    }
}

template <int N>
__global__ void scan_phaseC(const float* __restrict__ xsT,
                            const float* __restrict__ deltaT,
                            const float* __restrict__ xdbl,
                            const float* __restrict__ Abar,
                            const float* __restrict__ chunkH,
                            const float* __restrict__ Ds,
                            float* __restrict__ yT) {
    const int c = RR + 2 * N;
    int chunk = blockIdx.x;
    int bk    = blockIdx.y;
    int dd    = threadIdx.x;
    int k     = bk & 3;

    float Areg[N], h[N];
    size_t base = ((size_t)(chunk * (BB * KK) + bk) * DD + dd) * N;
#pragma unroll
    for (int i = 0; i < N; ++i) {
        Areg[i] = Abar[((size_t)k * DD + dd) * N + i];
        h[i]    = chunkH[base + i];
    }
    float Dv = Ds[k * DD + dd];

    const float* dtp = deltaT + (size_t)bk * LL * DD;
    const float* up  = xsT    + (size_t)bk * LL * DD;
    const float* xr  = xdbl   + (size_t)bk * LL * c;
    float*       yp  = yT     + (size_t)bk * LL * DD;

    int l0 = chunk * CLEN;
    for (int l = l0; l < l0 + CLEN; ++l) {
        float dt = dtp[(size_t)l * DD + dd];
        float u  = up[(size_t)l * DD + dd];
        float du = dt * u;
        const float* row = xr + (size_t)l * c;
        float y = u * Dv;
#pragma unroll
        for (int i = 0; i < N; ++i) {
            float dA = __expf(dt * Areg[i]);
            h[i] = dA * h[i] + du * row[RR + i];
            y   += h[i] * row[RR + N + i];
        }
        yp[(size_t)l * DD + dd] = y;
    }
}

// =====================================================================
// Combine 4 directions (with flips / spatial transpose) + LayerNorm(d)
// + multiply by SiLU(z). One block per (b,l), 192 threads.
// =====================================================================
__global__ void combine_ln(const float* __restrict__ yT,
                           const float* __restrict__ xz,
                           const float* __restrict__ ln_w,
                           const float* __restrict__ ln_b,
                           float* __restrict__ ynorm) {
    __shared__ float s1[DD], s2[DD], stats[2];
    int bl = blockIdx.x;
    int b  = bl / LL;
    int l  = bl % LL;
    int dd = threadIdx.x;
    int h  = l / WW, w = l % WW;
    int lt = w * HH + h;

    size_t base = (size_t)b * KK * LL * DD;
    float y = yT[base + (size_t)l * DD + dd]                        // dir 0
            + yT[base + (size_t)(2 * LL + (LL - 1 - l)) * DD + dd]  // flip(dir 2)
            + yT[base + (size_t)(LL + lt) * DD + dd]                // transp(dir 1)
            + yT[base + (size_t)(3 * LL + (LL - 1 - lt)) * DD + dd];// transp(flip(dir3))

    s1[dd] = y;
    s2[dd] = y * y;
    __syncthreads();
    for (int s = DD / 2; s >= 6; s >>= 1) {
        if (dd < s) { s1[dd] += s1[dd + s]; s2[dd] += s2[dd + s]; }
        __syncthreads();
    }
    if (dd == 0) {
        float a = 0.f, q = 0.f;
        for (int i = 0; i < 6; ++i) { a += s1[i]; q += s2[i]; }
        float mu  = a / (float)DD;
        float var = q / (float)DD - mu * mu;
        stats[0] = mu;
        stats[1] = rsqrtf(var + 1e-5f);
    }
    __syncthreads();
    float yn = (y - stats[0]) * stats[1] * ln_w[dd] + ln_b[dd];
    float z  = xz[((size_t)b * LL + l) * (2 * DD) + DD + dd];
    float sz = z / (1.f + __expf(-z));
    ynorm[((size_t)b * LL + l) * DD + dd] = yn * sz;
}

// =====================================================================
// Launch
// =====================================================================
extern "C" void kernel_launch(void* const* d_in, const int* in_sizes, int n_in,
                              void* d_out, int out_size, void* d_ws, size_t ws_size,
                              hipStream_t stream) {
    (void)in_sizes; (void)n_in; (void)out_size; (void)ws_size;
    const float* x    = (const float*)d_in[0];
    const float* g_w1 = (const float*)d_in[1];
    const float* g_b1 = (const float*)d_in[2];
    const float* g_w2 = (const float*)d_in[3];
    const float* g_b2 = (const float*)d_in[4];
    // per-branch params: in_w, conv_w, conv_b, xproj_w, dt_w, dt_b, A_logs, Ds, ln_w, ln_b, out_w
    const float* P[3][11];
    for (int i = 0; i < 3; ++i)
        for (int j = 0; j < 11; ++j)
            P[i][j] = (const float*)d_in[5 + 11 * i + j];

    const int M   = BB * LL;                 // 4608
    const int CMX = RR + 2 * 32;             // 70 (max c)

    // ---- workspace carve-up (floats) ----
    float* ws = (float*)d_ws;
    size_t off = 0;
    float* gates  = ws + off; off += 16;
    float* xz     = ws + off; off += (size_t)M * 2 * DD;            // 1.77M
    float* xsT    = ws + off; off += (size_t)BB * KK * LL * DD;     // 3.54M
    float* xdbl   = ws + off; off += (size_t)BB * KK * LL * CMX;    // 1.29M
    float* deltaT = ws + off; off += (size_t)BB * KK * LL * DD;     // 3.54M
    float* yT     = ws + off; off += (size_t)BB * KK * LL * DD;     // 3.54M
    float* ynorm  = ws + off; off += (size_t)M * DD;                // 0.88M
    float* Abar   = ws + off; off += (size_t)KK * DD * 32;          // 24K
    float* chunkA = ws + off; off += (size_t)NCHUNK * BB * KK * DD * 32;  // 2.36M
    float* chunkH = ws + off; off += (size_t)NCHUNK * BB * KK * DD * 32;  // 2.36M

    gate_kernel<<<1, 256, 0, stream>>>(x, g_w1, g_b1, g_w2, g_b2, gates);
    zero_kernel<<<(M * CC + 255) / 256, 256, 0, stream>>>((float*)d_out, M * CC);

    for (int br = 0; br < 3; ++br) {
        const int n = (br == 0) ? 8 : (br == 1) ? 16 : 32;
        const int c = RR + 2 * n;
        const float* in_w    = P[br][0];
        const float* conv_w  = P[br][1];
        const float* conv_b  = P[br][2];
        const float* xproj_w = P[br][3];
        const float* dt_w    = P[br][4];
        const float* dt_b    = P[br][5];
        const float* A_logs  = P[br][6];
        const float* Ds      = P[br][7];
        const float* ln_w    = P[br][8];
        const float* ln_b    = P[br][9];
        const float* out_w   = P[br][10];

        // xz = x @ in_w^T : (4608 x 384), K=96, N multiple of 16
        wmma_gemm_xwT<96, false, false>
            <<<dim3(2 * DD / 16, M / 16, 1), 32, 0, stream>>>(
                x, in_w, xz, 2 * DD, 0, 0, 0, 1, nullptr, 0);

        // depthwise conv + SiLU + cross-scan scatter
        conv_silu_scatter<<<(BB * DD * LL + 255) / 256, 256, 0, stream>>>(
            xz, conv_w, conv_b, xsT);

        // x_dbl[b,k] = xsT[b,k] (L x 192) @ xproj_w[k]^T (c x 192), batched over 8
        wmma_gemm_xwT<192, true, false>
            <<<dim3((c + 15) / 16, LL / 16, BB * KK), 32, 0, stream>>>(
                xsT, xproj_w, xdbl, c,
                (long)LL * DD, (long)c * DD, (long)LL * c, KK, nullptr, 0);

        // delta = softplus(dts @ dt_w^T + dt_b)
        dt_kernel<<<(BB * KK * LL * DD + 255) / 256, 256, 0, stream>>>(
            xdbl, dt_w, dt_b, deltaT, c);

        // Abar = -exp(A_logs)
        abar_kernel<<<(KK * DD * n + 255) / 256, 256, 0, stream>>>(
            A_logs, Abar, KK * DD * n);

        // chunked associative selective scan
        dim3 sg(NCHUNK, BB * KK);
        int totB = BB * KK * DD * n;
        if (n == 8) {
            scan_phaseA<8><<<sg, DD, 0, stream>>>(xsT, deltaT, xdbl, Abar, chunkA, chunkH);
            scan_phaseB<<<(totB + 255) / 256, 256, 0, stream>>>(chunkA, chunkH, totB);
            scan_phaseC<8><<<sg, DD, 0, stream>>>(xsT, deltaT, xdbl, Abar, chunkH, Ds, yT);
        } else if (n == 16) {
            scan_phaseA<16><<<sg, DD, 0, stream>>>(xsT, deltaT, xdbl, Abar, chunkA, chunkH);
            scan_phaseB<<<(totB + 255) / 256, 256, 0, stream>>>(chunkA, chunkH, totB);
            scan_phaseC<16><<<sg, DD, 0, stream>>>(xsT, deltaT, xdbl, Abar, chunkH, Ds, yT);
        } else {
            scan_phaseA<32><<<sg, DD, 0, stream>>>(xsT, deltaT, xdbl, Abar, chunkA, chunkH);
            scan_phaseB<<<(totB + 255) / 256, 256, 0, stream>>>(chunkA, chunkH, totB);
            scan_phaseC<32><<<sg, DD, 0, stream>>>(xsT, deltaT, xdbl, Abar, chunkH, Ds, yT);
        }

        // combine directions + LayerNorm + SiLU(z) gating
        combine_ln<<<BB * LL, DD, 0, stream>>>(yT, xz, ln_w, ln_b, ynorm);

        // out += gate * ynorm @ out_w^T : (4608 x 96), K=192, N multiple of 16
        wmma_gemm_xwT<192, false, true>
            <<<dim3(CC / 16, M / 16, 1), 32, 0, stream>>>(
                ynorm, out_w, (float*)d_out, CC, 0, 0, 0, 1, gates, br);
    }
}
